// SRUCell_53970559041917
// MI455X (gfx1250) — matrix operand order repaired and checked by reading, not compiled, not run on hardware
//
#include <hip/hip_runtime.h>
#include <stdint.h>

typedef __bf16 bf16t;
typedef __attribute__((ext_vector_type(16))) __bf16 v16bf;
typedef __attribute__((ext_vector_type(8)))  float  v8f;

#define B_   32
#define T_   1024
#define D_   1024
#define U_   1024
#define N4_  4096

#define MT 128
#define NT 128
#define KT 64
#define LDSK 80                        // bf16 row stride in LDS (160B, 16B-aligned)
#define STAGE_E (2 * MT * LDSK)        // bf16 elements per stage (A tile + B tile)
#define SMEM_BYTES (2 * STAGE_E * 2)   // double buffered: 81920 B
#define NSTAGE (D_ / KT)               // 16
#define CLD 136                        // epilogue bf16 row stride (272B, 16B-aligned)

// ---------- bf16 bit helpers (avoid relying on __bf16 conversions) ----------
__device__ __forceinline__ unsigned short f2bfbits(float f) {
  unsigned u = __float_as_uint(f);
  return (unsigned short)((u + 0x7FFFu + ((u >> 16) & 1u)) >> 16);   // RNE
}
__device__ __forceinline__ float bfbits2f(unsigned short s) {
  return __uint_as_float(((unsigned)s) << 16);
}

// ---------- convert x chunk [B][Tc][D] fp32 -> bf16 bits ----------
__global__ void k_cvt_x(const float* __restrict__ x, unsigned short* __restrict__ xb,
                        int t0, int Tc) {
  int idx = blockIdx.x * blockDim.x + threadIdx.x;   // over B*Tc*(D/4) float4s
  int per_b = Tc * (D_ / 4);
  if (idx >= B_ * per_b) return;
  int b   = idx / per_b;
  int rem = idx - b * per_b;
  const float4 v = ((const float4*)(x + (size_t)b * T_ * D_ + (size_t)t0 * D_))[rem];
  union { unsigned short h[4]; unsigned long long u; } o;
  o.h[0] = f2bfbits(v.x); o.h[1] = f2bfbits(v.y);
  o.h[2] = f2bfbits(v.z); o.h[3] = f2bfbits(v.w);
  ((unsigned long long*)xb)[idx] = o.u;
}

// ---------- transpose + convert W [D][4U] fp32 -> Wt [4U][D] bf16 bits ----------
__global__ void k_cvt_wt(const float* __restrict__ W, unsigned short* __restrict__ Wt) {
  __shared__ float tile[32][33];
  int tx = threadIdx.x, ty = threadIdx.y;            // 32 x 8
  int n0 = blockIdx.x * 32, k0 = blockIdx.y * 32;
#pragma unroll
  for (int j = 0; j < 32; j += 8)
    tile[ty + j][tx] = W[(size_t)(k0 + ty + j) * N4_ + n0 + tx];
  __syncthreads();
#pragma unroll
  for (int j = 0; j < 32; j += 8)
    Wt[(size_t)(n0 + ty + j) * D_ + k0 + tx] = f2bfbits(tile[tx][ty + j]);
}

// ---------- GEMM: G[b*Tc+t][n] = sum_k A[row][k] * Wt[n][k], bf16 in/out, f32 acc ----------
__global__ __launch_bounds__(256)
void k_gemm(const unsigned short* __restrict__ A,   // [B*Tc][1024] bf16 bits
            const unsigned short* __restrict__ Wt,  // [4096][1024] bf16 bits
            unsigned short* __restrict__ G,         // [B*Tc][4096] bf16 bits
            int Tc) {
  extern __shared__ char smem_raw[];
  bf16t* smem = (bf16t*)smem_raw;

  const int tid   = threadIdx.x;
  const int lane  = tid & 31;
  const int wave  = tid >> 5;
  const int waveM = wave & 3;
  const int waveN = wave >> 2;
  const int b       = blockIdx.z;
  const int rowBase = b * Tc + blockIdx.y * MT;
  const int colBase = blockIdx.x * NT;

  v8f zero = {0.f, 0.f, 0.f, 0.f, 0.f, 0.f, 0.f, 0.f};
  v8f acc[2][4];
#pragma unroll
  for (int mi = 0; mi < 2; ++mi)
#pragma unroll
    for (int ni = 0; ni < 4; ++ni) acc[mi][ni] = zero;

  // ---- per-thread async-copy addressing: 4 A chunks + 4 B chunks (16B each) per stage
  const int lrow = tid >> 3;          // 0..31
  const int c8   = tid & 7;           // 16B chunk within 64-K slab
  unsigned long long aAddr[4], bAddr[4];
  unsigned ldsA[4], ldsB[4];
  const unsigned sbase = (unsigned)(unsigned long long)(void*)smem_raw;
#pragma unroll
  for (int i = 0; i < 4; ++i) {
    int row = lrow + i * 32;
    aAddr[i] = (unsigned long long)(A  + (size_t)(rowBase + row) * D_ + c8 * 8);
    bAddr[i] = (unsigned long long)(Wt + (size_t)(colBase + row) * D_ + c8 * 8);
    ldsA[i]  = sbase + (unsigned)(row * LDSK + c8 * 8) * 2u;
    ldsB[i]  = sbase + (unsigned)(MT * LDSK + row * LDSK + c8 * 8) * 2u;
  }

  // ---- prologue: async-DMA stage 0 into buffer 0
#pragma unroll
  for (int i = 0; i < 4; ++i) {
    asm volatile("global_load_async_to_lds_b128 %0, %1, off"
                 :: "v"(ldsA[i]), "v"(aAddr[i]) : "memory");
    asm volatile("global_load_async_to_lds_b128 %0, %1, off"
                 :: "v"(ldsB[i]), "v"(bAddr[i]) : "memory");
    aAddr[i] += KT * 2;  bAddr[i] += KT * 2;
  }

  for (int s = 0; s < NSTAGE; ++s) {
    const int p = s & 1;
    if (s + 1 < NSTAGE) {
      const unsigned bofs = p ? 0u : (unsigned)(STAGE_E * 2);   // next buffer = 1-p
#pragma unroll
      for (int i = 0; i < 4; ++i) {
        asm volatile("global_load_async_to_lds_b128 %0, %1, off"
                     :: "v"(ldsA[i] + bofs), "v"(aAddr[i]) : "memory");
        asm volatile("global_load_async_to_lds_b128 %0, %1, off"
                     :: "v"(ldsB[i] + bofs), "v"(bAddr[i]) : "memory");
        aAddr[i] += KT * 2;  bAddr[i] += KT * 2;
      }
      // async loads retire in order: <=8 outstanding means stage s's 8 copies landed
      asm volatile("s_wait_asynccnt 0x8" ::: "memory");
    } else {
      asm volatile("s_wait_asynccnt 0x0" ::: "memory");
    }
    __syncthreads();                                   // stage-s tiles visible WG-wide

    const bf16t* Ab = smem + (size_t)p * STAGE_E;
    const bf16t* Bb = Ab + MT * LDSK;
    const unsigned int* As32 = (const unsigned int*)Ab;
    const unsigned int* Bs32 = (const unsigned int*)Bb;
#pragma unroll
    for (int kk = 0; kk < 2; ++kk) {
      v16bf af[2];
      {
        int arow = waveM * 32 + (lane & 15);
        int kbA  = kk * 32 + ((lane >> 4) << 3);
#pragma unroll
        for (int mi = 0; mi < 2; ++mi) {
          int base = (arow + mi * 16) * (LDSK / 2) + (kbA >> 1);
          union { v16bf v; uint4 q[2]; } t;
          t.q[0] = *(const uint4*)(As32 + base);       // K kb..kb+7
          t.q[1] = *(const uint4*)(As32 + base + 8);   // K kb+16..kb+23
          af[mi] = t.v;
        }
      }
      v16bf bfr[4];
      {
        int brow = waveN * 64 + (lane & 15);
        int kbB  = kk * 32 + ((lane >> 4) << 3);
#pragma unroll
        for (int ni = 0; ni < 4; ++ni) {
          int base = (brow + ni * 16) * (LDSK / 2) + (kbB >> 1);
          union { v16bf v; uint4 q[2]; } t;
          t.q[0] = *(const uint4*)(Bs32 + base);
          t.q[1] = *(const uint4*)(Bs32 + base + 8);
          bfr[ni] = t.v;
        }
      }
#pragma unroll
      for (int mi = 0; mi < 2; ++mi)
#pragma unroll
        for (int ni = 0; ni < 4; ++ni)
          acc[mi][ni] = __builtin_amdgcn_wmma_f32_16x16x32_bf16(
              false, af[mi], false, bfr[ni], (short)0, acc[mi][ni], false, false);
    }
    __syncthreads();                                   // buf p free for stage s+2 DMA
  }

  // ---- epilogue: pack accumulators to bf16 in LDS, then coalesced b128 stores
  unsigned short* Cs = (unsigned short*)smem_raw;
  {
    const int crow0 = waveM * 32;
    const int ccol0 = waveN * 64 + (lane & 15);
    const int rofs  = (lane >> 4) * 8;
#pragma unroll
    for (int mi = 0; mi < 2; ++mi)
#pragma unroll
      for (int ni = 0; ni < 4; ++ni)
#pragma unroll
        for (int v = 0; v < 8; ++v)
          Cs[(crow0 + mi * 16 + v + rofs) * CLD + ccol0 + ni * 16] =
              f2bfbits(acc[mi][ni][v]);
  }
  __syncthreads();
  // 128x128 bf16 tile = 2048 uint4 -> 8 per thread
#pragma unroll
  for (int i = 0; i < 8; ++i) {
    int idx = tid + i * 256;
    int row = idx >> 4;                 // 0..127
    int q   = idx & 15;                 // uint4 within row
    uint4 val = *(const uint4*)(Cs + row * CLD + q * 8);
    uint4* dst = (uint4*)(G + (size_t)(b * Tc + blockIdx.y * MT + row) * N4_ + colBase);
    dst[q] = val;
  }
}

// ---------- sequential SRU scan over a T chunk (software-pipelined) ----------
__global__ __launch_bounds__(256)
void k_scan(const unsigned short* __restrict__ g, const float* __restrict__ bias,
            const float* __restrict__ Vr, const float* __restrict__ Vf,
            const float* __restrict__ cin, float* __restrict__ cout,
            float* __restrict__ h, int Tc, int t0) {
  int gid = blockIdx.x * blockDim.x + threadIdx.x;   // b*U + u
  int u = gid & (U_ - 1);
  int b = gid >> 10;
  float c  = cin[gid];
  float vf = Vf[u], vr = Vr[u];
  float b1 = bias[u], b2 = bias[U_ + u], b3 = bias[2 * U_ + u], b4 = bias[3 * U_ + u];
  const unsigned short* gb = g + (size_t)b * Tc * N4_;
  float* hb = h + ((size_t)b * T_ + t0) * U_;

  // prefetch t=0 gate values; loads for future t are independent of the carry c,
  // so the scheduler can keep multiple iterations of loads in flight
  unsigned short n1 = gb[u], n2 = gb[U_ + u], n3 = gb[2 * U_ + u], n4 = gb[3 * U_ + u];
#pragma unroll 4
  for (int t = 0; t < Tc; ++t) {
    float a1 = bfbits2f(n1) + b1;
    float a2 = bfbits2f(n2) + b2;
    float a3 = bfbits2f(n3) + b3;
    float a4 = bfbits2f(n4) + b4;
    if (t + 1 < Tc) {
      const unsigned short* nrow = gb + (size_t)(t + 1) * N4_;
      n1 = nrow[u]; n2 = nrow[U_ + u]; n3 = nrow[2 * U_ + u]; n4 = nrow[3 * U_ + u];
    }
    float f  = 1.f / (1.f + __expf(-(a1 + vf * c)));
    float cn = f * c + (1.f - f) * a2;
    float r  = 1.f / (1.f + __expf(-(a3 + vr * c)));   // uses old c
    c = cn;
    hb[(size_t)t * U_ + u] = r * cn + (1.f - r) * a4;
  }
  cout[gid] = c;
}

// ---------- host ----------
extern "C" void kernel_launch(void* const* d_in, const int* in_sizes, int n_in,
                              void* d_out, int out_size, void* d_ws, size_t ws_size,
                              hipStream_t stream) {
  (void)in_sizes; (void)n_in; (void)out_size;
  const float* x    = (const float*)d_in[0];
  // d_in[1] = h0 (unused by the reference computation)
  const float* c0   = (const float*)d_in[2];
  const float* W    = (const float*)d_in[3];
  const float* bias = (const float*)d_in[4];
  const float* Vr   = (const float*)d_in[5];
  const float* Vf   = (const float*)d_in[6];
  float* h = (float*)d_out;

  char* ws = (char*)d_ws;
  size_t off = 0;
  unsigned short* Wt = (unsigned short*)(ws + off);  off += (size_t)N4_ * D_ * 2;   // 8 MB
  float* cstate = (float*)(ws + off);                off += (size_t)B_ * U_ * 4;    // 128 KB

  // Pick largest T chunk that fits in the workspace (min 128 for the 128-row GEMM tile).
  int Tc = 128;
  for (int cand = 1024; cand >= 128; cand >>= 1) {
    size_t need = off + (size_t)B_ * cand * D_ * 2 + (size_t)B_ * cand * N4_ * 2 + 1024;
    if (need <= ws_size) { Tc = cand; break; }
  }
  unsigned short* xb = (unsigned short*)(ws + off);
  unsigned short* g  = (unsigned short*)(ws + off + (size_t)B_ * Tc * D_ * 2);

  k_cvt_wt<<<dim3(N4_ / 32, D_ / 32), dim3(32, 8), 0, stream>>>(W, Wt);

  for (int t0 = 0; t0 < T_; t0 += Tc) {
    int nquad = B_ * Tc * (D_ / 4);
    k_cvt_x<<<(nquad + 255) / 256, 256, 0, stream>>>(x, xb, t0, Tc);
    k_gemm<<<dim3(N4_ / NT, Tc / MT, B_), 256, SMEM_BYTES, stream>>>(xb, Wt, g, Tc);
    k_scan<<<(B_ * U_) / 256, 256, 0, stream>>>(g, bias, Vr, Vf,
                                                (t0 == 0) ? c0 : cstate, cstate,
                                                h, Tc, t0);
  }
}
